// EEG_KAN_12506944766057
// MI455X (gfx1250) — matrix operand order repaired and checked by reading, not compiled, not run on hardware
//
#include <hip/hip_runtime.h>
#include <hip/hip_bf16.h>

typedef __attribute__((ext_vector_type(16))) _Float16 v16h;
typedef __attribute__((ext_vector_type(8)))  _Float16 v8h;
typedef __attribute__((ext_vector_type(8)))  float    v8f;

#define KAN_BATCH 16384

// ---------------------------------------------------------------------------
// Kernel 1: pack base_w [n_out, n_in] + spline_w [n_out, n_in, 8] into the
// exact v_wmma_f32_16x16x32_f16 B-fragment layout:
//   element j (0..15) of lane L of k-block kb, n-block nb holds
//   W_exp[K = kb*32 + 16*(j>=8) + 8*(L>=16) + (j&7), N = nb*16 + (L&15)]
// where W_exp[i*9+0, o] = base_w[o,i], W_exp[i*9+1+g, o] = spline_w[o,i,g].
// Zero-padded for K >= n_in*9 and N >= n_out, so the GEMM needs no K/N masks
// (Ntiles may be padded beyond ceil(n_out/16) for an even group split).
// ---------------------------------------------------------------------------
__global__ void kan_pack_w(const float* __restrict__ base_w,
                           const float* __restrict__ spline_w,
                           _Float16* __restrict__ Wp,
                           int n_in, int n_out, int Ktot, int Ntiles, int total)
{
    int tid = blockIdx.x * blockDim.x + threadIdx.x;
    if (tid >= total) return;
    int j    = tid & 15;
    int lane = (tid >> 4) & 31;
    int q    = tid >> 9;               // kb*Ntiles + nb
    int nb   = q % Ntiles;
    int kb   = q / Ntiles;
    int Kg   = kb * 32 + ((j >> 3) << 4) + ((lane >> 4) << 3) + (j & 7);
    int Ng   = nb * 16 + (lane & 15);
    float v = 0.0f;
    if (Ng < n_out && Kg < Ktot) {
        int i = Kg / 9;
        int r = Kg - 9 * i;
        v = (r == 0) ? base_w[(size_t)Ng * n_in + i]
                     : spline_w[((size_t)Ng * n_in + i) * 8 + (r - 1)];
    }
    Wp[tid] = (_Float16)v;
}

// ---------------------------------------------------------------------------
// Kernel 2: expand activations.  For each (row, i) compute silu(x) and the
// 8 order-3 B-spline basis values on the fixed uniform grid
// t_p = 0.4f*(p-3) - 1 (p = 0..11), exactly mirroring the reference's
// Cox-de Boor recursion (knots constant-fold; divisions become reciprocal
// multiplies).  Writes 9 f16 features at A[row*Kpad + i*9 + {0..8}].
// Thread handling i == n_in-1 also zeroes the K-pad tail of its row.
// ---------------------------------------------------------------------------
__global__ void kan_expand(const float* __restrict__ H,
                           _Float16* __restrict__ A,
                           int n_in, int Ktot, int Kpad)
{
    int tid = blockIdx.x * blockDim.x + threadIdx.x;
    int total = KAN_BATCH * n_in;
    if (tid >= total) return;
    int row = tid / n_in;
    int i   = tid - row * n_in;

    float x = H[(size_t)row * n_in + i];

    float t[12];
#pragma unroll
    for (int p = 0; p < 12; ++p) t[p] = 0.4f * (float)(p - 3) - 1.0f;

    float bs[11];
#pragma unroll
    for (int p = 0; p < 11; ++p)
        bs[p] = (x >= t[p] && x < t[p + 1]) ? 1.0f : 0.0f;

#pragma unroll
    for (int d = 1; d <= 3; ++d) {
#pragma unroll
        for (int p = 0; p + d < 11; ++p) {
            float left  = (x - t[p]) / (t[p + d] - t[p]) * bs[p];
            float right = (t[p + d + 1] - x) / (t[p + d + 1] - t[p + 1]) * bs[p + 1];
            bs[p] = left + right;
        }
    }

    float s = x / (1.0f + __expf(-x));   // silu residual branch

    _Float16* dst = A + (size_t)row * Kpad + (size_t)i * 9;
    dst[0] = (_Float16)s;
#pragma unroll
    for (int g = 0; g < 8; ++g) dst[1 + g] = (_Float16)bs[g];

    if (i == n_in - 1) {
        _Float16* rowp = A + (size_t)row * Kpad;
        for (int k = Ktot; k < Kpad; ++k) rowp[k] = (_Float16)0.0f;
    }
}

// ---------------------------------------------------------------------------
// Kernel 3: register-resident, double-buffered WMMA GEMM.
// C[B, n_out] = A[B, Kpad] x W_exp.  256 threads = 8 wave32s; wave w of
// block (bx, by) owns rows 16*(8*bx+w)..+15 and column tiles
// [by*NT, (by+1)*NT).  Per 32-wide K step kb:
//   - the A fragment (two contiguous global_load_b128, no transpose needed)
//     and ALL NT B fragments (one 32B load each from the pre-swizzled,
//     L2-resident Wp) for step kb+1 are issued as one batch FIRST,
//   - then the NT WMMAs of step kb run against the previous register set,
//     so every WMMA waits only on loads issued a full K-step earlier
//     (s_wait_loadcnt tolerates ~2*NT in-flight loads; latency fully hidden).
// The one-step-ahead loads at kb = Kblocks-1 are dead but stay inside the
// workspace (<=64B past A's end lands in the Wp slot; <=12KB past the packed
// weights stays inside the 2MB Wp slot), so no clamping -> pure pointer
// increments, no cselect/mul chains in the loop.
// No LDS, no barriers, no EXEC masking anywhere near the WMMAs.
// ---------------------------------------------------------------------------
template <int NT>
__global__ __launch_bounds__(256)
void kan_wmma_gemm(const _Float16* __restrict__ A,
                   const _Float16* __restrict__ Wp,
                   float* __restrict__ C,
                   int Kpad, int n_out, int Ntiles)
{
    const int lane = threadIdx.x & 31;
    const int wave = threadIdx.x >> 5;
    const int m0   = (blockIdx.x * 8 + wave) * 16;
    const int tg   = blockIdx.y * NT;          // first column tile of group

    const int Kblocks = Kpad >> 5;

    // per-lane A source: row m0 + (lane&15), first chunk at K = 8*(lane>>4)
    const _Float16* arow = A + (size_t)(m0 + (lane & 15)) * Kpad + ((lane >> 4) << 3);
    // per-lane B source: 32B fragment slice for this lane, group-offset
    const _Float16* bbase = Wp + ((size_t)tg * 32 + lane) * 16;
    const size_t bKstride = (size_t)Ntiles * 512;   // f16 elements per K block

    auto loadA = [&](int kb) -> v16h {
        const _Float16* ap = arow + (size_t)kb * 32;
        v8h lo = *(const v8h*)(ap);        // K = kbase .. kbase+7
        v8h hi = *(const v8h*)(ap + 16);   // K = kbase+16 .. kbase+23
        return __builtin_shufflevector(lo, hi,
                   0, 1, 2, 3, 4, 5, 6, 7, 8, 9, 10, 11, 12, 13, 14, 15);
    };
    auto loadB = [&](int kb, int t) -> v16h {
        return *(const v16h*)(bbase + (size_t)kb * bKstride + (size_t)t * 512);
    };

    v8f acc[NT];
#pragma unroll
    for (int t = 0; t < NT; ++t) acc[t] = (v8f){};

    // prologue: fill buffer 0
    v16h a = loadA(0);
    v16h bf[NT];
#pragma unroll
    for (int t = 0; t < NT; ++t) bf[t] = loadB(0, t);

    for (int kb = 0; kb < Kblocks; ++kb) {
        // ---- issue next K-step's loads first (dead on the last trip) ----
        v16h a_next = loadA(kb + 1);
        v16h bn[NT];
#pragma unroll
        for (int t = 0; t < NT; ++t) bn[t] = loadB(kb + 1, t);

        __builtin_prefetch(arow + (size_t)(kb + 8) * 32, 0, 3);  // stream A ahead

        // ---- math for the current K-step against the previous buffer ----
#pragma unroll
        for (int t = 0; t < NT; ++t)
            acc[t] = __builtin_amdgcn_wmma_f32_16x16x32_f16(
                         false, a, false, bf[t], (short)0, acc[t], false, false);

        a = a_next;
#pragma unroll
        for (int t = 0; t < NT; ++t) bf[t] = bn[t];
    }

    // D layout: VGPR r, lane l -> M = r + 8*(l>>4), N = l&15
    const int col   = lane & 15;
    const int rbase = (lane >> 4) << 3;
#pragma unroll
    for (int t = 0; t < NT; ++t) {
        int n = (tg + t) * 16 + col;
        if (n < n_out) {
#pragma unroll
            for (int r = 0; r < 8; ++r)
                C[(size_t)(m0 + rbase + r) * n_out + n] = acc[t][r];
        }
    }
}

// ---------------------------------------------------------------------------
// Host launcher: 4 layers x (pack_w -> expand -> gemm), all on `stream`.
// Workspace layout (bytes):
//   A  : f16 expanded activations, max 16384*2816*2 = 92,274,688
//   Wp : f16 swizzled weights,     max 2816*192*2  = 1,081,344 (2 MB slot)
//   H1 : f32 [16384,168] = 11,010,048   (reused for layer-2 output)
//   H2 : f32 [16384, 64] =  4,194,304
// Total ~110 MB.  Layer tiling: Ntiles padded to 12/4/12/1 so the column-tile
// groups split evenly (NT x groups = 6x2 / 2x2 / 6x2 / 1x1), doubling wave
// parallelism on the big layers for one extra L2-resident read of A.
// ---------------------------------------------------------------------------
extern "C" void kernel_launch(void* const* d_in, const int* in_sizes, int n_in_cnt,
                              void* d_out, int out_size, void* d_ws, size_t ws_size,
                              hipStream_t stream)
{
    (void)in_sizes; (void)n_in_cnt; (void)out_size; (void)ws_size;

    const float* x = (const float*)d_in[0];
    const float* base_w[4]   = { (const float*)d_in[2], (const float*)d_in[5],
                                 (const float*)d_in[8], (const float*)d_in[11] };
    const float* spline_w[4] = { (const float*)d_in[3], (const float*)d_in[6],
                                 (const float*)d_in[9], (const float*)d_in[12] };

    char* ws = (char*)d_ws;
    const size_t A_SZ  = (size_t)KAN_BATCH * 2816 * sizeof(_Float16); // 92,274,688
    const size_t W_SZ  = (size_t)2 << 20;
    const size_t H1_SZ = (size_t)KAN_BATCH * 168 * sizeof(float);

    _Float16* A  = (_Float16*)(ws);
    _Float16* Wp = (_Float16*)(ws + A_SZ);
    float*    H1 = (float*)(ws + A_SZ + W_SZ);
    float*    H2 = (float*)(ws + A_SZ + W_SZ + H1_SZ);

    const int n_in_l[4]  = { 310, 168, 64, 168 };
    const int n_out_l[4] = { 168, 64, 168, 3 };
    const int ntiles_l[4]= { 12,  4,   12,  1 };   // padded for even grouping
    const int nt_l[4]    = { 6,   2,   6,   1 };   // tiles per wave group
    const float* hin[4]  = { x,  H1, H2, H1 };
    float*       hout[4] = { H1, H2, H1, (float*)d_out };

    for (int l = 0; l < 4; ++l) {
        const int Ktot    = n_in_l[l] * 9;
        const int Kpad    = (Ktot + 31) & ~31;
        const int Kblocks = Kpad >> 5;
        const int Ntiles  = ntiles_l[l];
        const int groups  = Ntiles / nt_l[l];

        const int packTotal = Kblocks * Ntiles * 32 * 16;
        kan_pack_w<<<(packTotal + 255) / 256, 256, 0, stream>>>(
            base_w[l], spline_w[l], Wp, n_in_l[l], n_out_l[l], Ktot, Ntiles, packTotal);

        const int expTotal = KAN_BATCH * n_in_l[l];
        kan_expand<<<(expTotal + 255) / 256, 256, 0, stream>>>(
            hin[l], A, n_in_l[l], Ktot, Kpad);

        const dim3 ggrid(KAN_BATCH / 16 / 8, groups);   // (128, groups)
        switch (nt_l[l]) {
        case 6:
            kan_wmma_gemm<6><<<ggrid, 256, 0, stream>>>(A, Wp, hout[l], Kpad, n_out_l[l], Ntiles);
            break;
        case 2:
            kan_wmma_gemm<2><<<ggrid, 256, 0, stream>>>(A, Wp, hout[l], Kpad, n_out_l[l], Ntiles);
            break;
        default:
            kan_wmma_gemm<1><<<ggrid, 256, 0, stream>>>(A, Wp, hout[l], Kpad, n_out_l[l], Ntiles);
            break;
        }
    }
}
